// HardBinaryConv_72206990181170
// MI455X (gfx1250) — matrix operand order repaired and checked by reading, not compiled, not run on hardware
//
#include <hip/hip_runtime.h>
#include <hip/hip_bf16.h>
#include <stdint.h>

typedef __attribute__((ext_vector_type(16))) __bf16 v16bf;
typedef __attribute__((ext_vector_type(8)))  float  v8f;
typedef __attribute__((ext_vector_type(4)))  int    v4i;

typedef __attribute__((address_space(1))) v4i gv4i;   // global int4
typedef __attribute__((address_space(3))) v4i lv4i;   // LDS int4

#define C_OUT 256
#define C_IN  256
#define KDIM  2304      // C_IN * 9
#define H_    56
#define W_    56
#define HW_   3136
#define NTOT  100352    // 32 * 3136
#define KC    32        // K per WMMA step
#define KSTEPS 72       // KDIM / KC
#define MTILE 256
#define NTILE 128
#define LDA   40        // padded LDS row stride (uint16 units): 80B, conflict-free, 16B-aligned
#define LDB   40

#if __has_builtin(__builtin_amdgcn_global_load_async_to_lds_b128) && \
    __has_builtin(__builtin_amdgcn_s_wait_asynccnt)
#define USE_ASYNC_LDS 1
#else
#define USE_ASYNC_LDS 0
#endif

static __device__ __forceinline__ uint16_t f2bf(float f) {
    union { float f; uint32_t u; } v; v.f = f;
    uint32_t r = v.u + 0x7FFFu + ((v.u >> 16) & 1u);   // round-to-nearest-even
    return (uint16_t)(r >> 16);
}

static __device__ __forceinline__ uint32_t pack_bf16(float a, float b) {
#if __has_builtin(__builtin_amdgcn_cvt_pk_bf16_f32)
    typedef __attribute__((ext_vector_type(2))) __bf16 v2bf;
    union { v2bf v; uint32_t u; } u;
    u.v = __builtin_amdgcn_cvt_pk_bf16_f32(a, b);
    return u.u;
#else
    return (uint32_t)f2bf(a) | ((uint32_t)f2bf(b) << 16);
#endif
}

// ---------------- Kernel 1: per-channel scale + binarize + K-permute to bf16 ----------------
// GEMM K-order: k' = (kh*3+kw)*256 + cin   (so each 32-chunk has constant kh,kw)
__global__ __launch_bounds__(256) void binarize_w(const float* __restrict__ w,
                                                  uint16_t* __restrict__ wb) {
    __shared__ float red[256];
    const int c = blockIdx.x;
    const int t = threadIdx.x;
    const float* wr = w + c * KDIM;
    float s = 0.f;
    for (int i = t; i < KDIM; i += 256) s += fabsf(wr[i]);
    red[t] = s;
    __syncthreads();
    for (int off = 128; off > 0; off >>= 1) {
        if (t < off) red[t] += red[t + off];
        __syncthreads();
    }
    const float scale = red[0] * (1.0f / (float)KDIM);
    for (int i = t; i < KDIM; i += 256) {
        float v = wr[i];                 // i = cin*9 + r
        int cin = i / 9;
        int r   = i - cin * 9;
        wb[c * KDIM + r * 256 + cin] = f2bf((v >= 0.f) ? scale : -scale);
    }
}

// ---------------- Kernel 2: implicit-GEMM binary conv via bf16 WMMA ----------------
// Double-buffered LDS; A tile staged via GLOBAL_LOAD_ASYNC_TO_LDS (no VGPR staging).
__global__ __launch_bounds__(256) void bconv_wmma(const float* __restrict__ x,
                                                  const uint16_t* __restrict__ wb,
                                                  float* __restrict__ out) {
    __shared__ alignas(16) uint16_t sA[2][MTILE * LDA];  // 2 x 20480 B
    __shared__ alignas(16) uint16_t sB[2][NTILE * LDB];  // 2 x 10240 B

    const int t     = threadIdx.x;
    const int lane  = t & 31;
    const int l15   = lane & 15;
    const int hi    = lane >> 4;
    const int wid   = t >> 5;       // 0..7
    const int waveM = wid & 3;      // 4 waves along M
    const int waveN = wid >> 2;     // 2 waves along N
    const int pBase = blockIdx.x * NTILE;

    // im2col coordinates for this thread's B-tile work
    const int nloc  = t & 127;      // column within N tile
    const int kHalf = t >> 7;       // 0 or 1: which 16 K-rows of the chunk
    const int p     = pBase + nloc;
    const int nimg  = p / HW_;
    const int hw    = p - nimg * HW_;
    const int h     = hw / W_;
    const int wcol  = hw - h * W_;
    const float* xn = x + (size_t)nimg * (C_IN * HW_) + (size_t)(kHalf * 16) * HW_;

    const uint4* wbRow = (const uint4*)(wb + (size_t)t * KDIM);  // A row t (k' order)

    float bReg[16];   // only B is register-staged (16 VGPRs)

    // ---- A tile: global -> LDS, asynchronously (ASYNCcnt), zero VGPR staging
    auto issueA = [&](int kb, int buf) {
        const uint4* g = wbRow + kb * (KC / 8);
        uint16_t*    l = sA[buf] + t * LDA;
#if USE_ASYNC_LDS
#pragma unroll
        for (int i = 0; i < 4; ++i)
            __builtin_amdgcn_global_load_async_to_lds_b128(
                (gv4i*)(g + i), (lv4i*)(l + i * 8), 0, 0);
#else
        uint4* s = (uint4*)l;
        s[0] = g[0]; s[1] = g[1]; s[2] = g[2]; s[3] = g[3];
#endif
    };
    auto waitA = [&]() {
#if USE_ASYNC_LDS
        __builtin_amdgcn_s_wait_asynccnt(0);
#endif
    };

    // ---- B tile: im2col gather into regs; chunk kb has constant (kh,kw)
    auto loadB = [&](int kb) {
        const int r   = kb >> 3;            // 0..8 (uniform -> SALU)
        const int kh  = (r * 11) >> 5;      // r/3 for r<=8
        const int kw  = r - kh * 3;
        const int hin = h + kh - 1;
        const int win = wcol + kw - 1;
        const bool ok = ((unsigned)hin < (unsigned)H_) && ((unsigned)win < (unsigned)W_);
        const float* px = xn + (size_t)((kb & 7) * 32) * HW_ + hin * W_ + win;
#pragma unroll
        for (int j = 0; j < 16; ++j)
            bReg[j] = ok ? px[(size_t)j * HW_] : 0.f;
    };
    auto storeB = [&](int buf) {
        uint32_t* sb = (uint32_t*)(sB[buf] + nloc * LDB + kHalf * 16);
#pragma unroll
        for (int j = 0; j < 16; j += 2)
            sb[j >> 1] = pack_bf16(bReg[j], bReg[j + 1]);
    };

    const v8f vzero = {0.f,0.f,0.f,0.f,0.f,0.f,0.f,0.f};
    v8f acc[4][4];
#pragma unroll
    for (int i = 0; i < 4; ++i)
#pragma unroll
        for (int j = 0; j < 4; ++j) acc[i][j] = vzero;

    // prologue: stage K-chunk 0 into buffer 0
    issueA(0, 0);
    loadB(0);
    storeB(0);
    waitA();

    int cur = 0;
    for (int kb = 0; kb < KSTEPS; ++kb) {
        __syncthreads();                       // buf[cur] ready; buf[cur^1] free
        const int kn = (kb + 1 < KSTEPS) ? kb + 1 : kb;
        issueA(kn, cur ^ 1);                   // async engine fills next A buffer
        loadB(kn);                             // global loads overlap WMMA below

        union AF { v16bf v; uint4 q[2]; } a[4], b[4];
        const uint16_t* sAc = sA[cur];
        const uint16_t* sBc = sB[cur];
        const int mBase = waveM * 64;
        const int nBase = waveN * 64;
#pragma unroll
        for (int im = 0; im < 4; ++im) {
            // ISA 16-bit A layout: lanes0-15 K0..7 & K16..23 ; lanes16-31 K8..15 & K24..31
            const uint16_t* pa = sAc + (mBase + im * 16 + l15) * LDA + hi * 8;
            a[im].q[0] = *(const uint4*)pa;
            a[im].q[1] = *(const uint4*)(pa + 16);
        }
#pragma unroll
        for (int in = 0; in < 4; ++in) {
            // B stored [n][k]: lanes0-15 K0..7 & K16..23 ; lanes16-31 K8..15 & K24..31
            const uint16_t* pb = sBc + (nBase + in * 16 + l15) * LDB + hi * 8;
            b[in].q[0] = *(const uint4*)pb;
            b[in].q[1] = *(const uint4*)(pb + 16);
        }
#pragma unroll
        for (int im = 0; im < 4; ++im)
#pragma unroll
            for (int in = 0; in < 4; ++in)
                acc[im][in] = __builtin_amdgcn_wmma_f32_16x16x32_bf16(
                    false, a[im].v, false, b[in].v,
                    (short)0, acc[im][in], false, false);

        storeB(cur ^ 1);                       // fill the other B buffer
        waitA();                               // A copies landed before next barrier
        cur ^= 1;
    }

    // ---- write out: out[(n*C_OUT + co)*HW + hw] ; lanes 0-15 contiguous in hw
#pragma unroll
    for (int in = 0; in < 4; ++in) {
        const int pcol = pBase + waveN * 64 + in * 16 + l15;
        const int n2   = pcol / HW_;
        const int off  = pcol + n2 * (HW_ * (C_OUT - 1)); // = n2*C_OUT*HW + hw
#pragma unroll
        for (int im = 0; im < 4; ++im) {
            const int co = waveM * 64 + im * 16 + hi * 8;
            float* o = out + (size_t)co * HW_ + off;
#pragma unroll
            for (int r2 = 0; r2 < 8; ++r2)
                o[(size_t)r2 * HW_] = acc[im][in][r2];
        }
    }
}

extern "C" void kernel_launch(void* const* d_in, const int* in_sizes, int n_in,
                              void* d_out, int out_size, void* d_ws, size_t ws_size,
                              hipStream_t stream) {
    const float* x = (const float*)d_in[0];   // (32,256,56,56) fp32
    const float* w = (const float*)d_in[1];   // (256*256*9, 1) fp32
    float* out     = (float*)d_out;           // (32,256,56,56) fp32
    uint16_t* wb   = (uint16_t*)d_ws;         // bf16 binarized weights (K-permuted), 1.18 MB

    binarize_w<<<C_OUT, 256, 0, stream>>>(w, wb);
    bconv_wmma<<<NTOT / NTILE, 256, 0, stream>>>(x, wb, out);
}
